// SelectiveScanSSM_22952305230024
// MI455X (gfx1250) — compile-verified, matrix operand order checked by reading, not run on hardware
//
#include <hip/hip_runtime.h>
#include <hip/hip_bf16.h>

// ---------------------------------------------------------------------------
// SelectiveScanSSM for MI455X (gfx1250, wave32).
// The three big GEMMs (86 GFLOP) run on v_wmma_f32_16x16x32_bf16 with the
// Tensor Data Mover staging 64x32 bf16 tiles into LDS (double-buffered,
// TENSORcnt-tracked). Everything else is bandwidth-trivial VALU work.
// ---------------------------------------------------------------------------

typedef __attribute__((ext_vector_type(16))) __bf16 v16bf;
typedef __attribute__((ext_vector_type(8)))  __bf16 v8bf;
typedef __attribute__((ext_vector_type(8)))  float  v8f;
typedef __attribute__((ext_vector_type(4)))  unsigned int u32x4;
typedef __attribute__((ext_vector_type(8)))  int i32x8;
typedef __attribute__((ext_vector_type(4)))  int i32x4;

// Problem constants (from the reference setup).
#define BSZ   2
#define SEQL  2048
#define DM    1024
#define DI    2048
#define NST   16
#define KCONV 4
#define ROWS  (BSZ * SEQL)     // 4096

// ---------------------------------------------------------------------------
// fp32 -> bf16 cast
// ---------------------------------------------------------------------------
__global__ void cast_f32_bf16(const float* __restrict__ in,
                              __bf16* __restrict__ out, size_t n) {
  size_t i = (size_t)blockIdx.x * blockDim.x + threadIdx.x;
  if (i < n) out[i] = (__bf16)in[i];
}

// ---------------------------------------------------------------------------
// TDM: DMA one 64-row x 32-col bf16 tile (row stride = k_len elements) from
// global memory into LDS at byte offset lds_off. Tile rows land contiguously
// (64 B per row). D# layout per CDNA5 ISA 8.3/8.4; groups 2/3 zero => dims
// >=2 unused (2D tensor). 6-arg builtin form (clang-23 / therock-10.0).
// ---------------------------------------------------------------------------
__device__ __forceinline__ void tdm_load_tile64x32(unsigned lds_off,
                                                   const __bf16* gptr,
                                                   unsigned k_len) {
  unsigned long long ga = (unsigned long long)(size_t)gptr;
  u32x4 g0;
  g0[0] = 0x1u;                                   // count=1, user descriptor
  g0[1] = lds_off;                                // lds_addr (bytes)
  g0[2] = (unsigned)ga;                           // global_addr[31:0]
  g0[3] = (unsigned)((ga >> 32) & 0x01FFFFFFu)    // global_addr[56:32]
        | (2u << 30);                             // type=2 ("image")

  i32x8 g1;
  unsigned td0 = k_len;                           // tensor_dim0 (elements)
  unsigned td1 = 64u;                             // tensor_dim1 (rows)
  g1[0] = (int)(1u << 16);                        // data_size=1 -> 2 bytes
  g1[1] = (int)((td0 & 0xFFFFu) << 16);           // tensor_dim0[15:0]
  g1[2] = (int)((td0 >> 16) | ((td1 & 0xFFFFu) << 16));
  g1[3] = (int)((td1 >> 16) | (32u << 16));       // tile_dim0 = 32
  g1[4] = (int)64u;                               // tile_dim1 = 64, tile_dim2=0
  g1[5] = (int)k_len;                             // tensor_dim0_stride[31:0]
  g1[6] = 0;                                      // stride[47:32]=0, dim1_stride lo=0
  g1[7] = 0;

  i32x4 gz4  = {0, 0, 0, 0};
  i32x8 gz8  = {0, 0, 0, 0, 0, 0, 0, 0};
  __builtin_amdgcn_tensor_load_to_lds(g0, g1, gz4, gz4, gz8, 0);
}

// The TDM builtin is not modeled as writing the LDS global. With no visible
// stores AND no escapes, the compiler proves the LDS contents undef and
// deletes the fragment loads. Escaping the LDS pointer into a memory-clobber
// asm gives the global a non-load use and a potential writer, so the loads
// must be materialized — while provenance stays intact, so the accesses still
// lower to ds ops (no volatile: that would force serialized SYS-scope flat).
__device__ __forceinline__ void tdm_publish_fence(const void* lds_base) {
  asm volatile("" :: "v"(lds_base) : "memory");
}

// ---------------------------------------------------------------------------
// WMMA bf16 GEMM:  C[M x Nc] (f32, row-major) = A[M x K] * B[Nc x K]^T
// Block = 128 threads = 4 waves, block tile 64(M) x 64(N); wave (2x2) owns a
// 32x32 C region (4 accumulators; 8 ds_load_b128 : 4 wmma per K step).
// LDS: double-buffered A/B tiles (4 KB each, 16 KB total), filled by the TDM.
// Fragment lane layout per ISA 16-bit A-matrix table:
//   lane<16 : row=lane,    halves 0..7 -> K+0..7,  halves 8..15 -> K+16..23
//   lane>=16: row=lane-16, halves 0..7 -> K+8..15, halves 8..15 -> K+24..31
// In the 64B LDS row that is bytes [kh*16, +16) and [32+kh*16, +16).
// ---------------------------------------------------------------------------
__device__ __forceinline__ v16bf lds_frag(const char* p) {
  union { v16bf v; v8bf h[2]; } u;
  u.h[0] = *(const v8bf*)(p);
  u.h[1] = *(const v8bf*)(p + 32);
  return u.v;
}

__global__ void __launch_bounds__(128)
gemm_bf16_tdm(const __bf16* __restrict__ A,
              const __bf16* __restrict__ Bm,
              float* __restrict__ C,
              int M, int Nc, int K) {
  __shared__ __align__(16) char smem[16384];  // A0 A1 B0 B1, 4 KB each

  const int lane = threadIdx.x & 31;
  const int wave = threadIdx.x >> 5;
  const int m0   = blockIdx.y * 64;
  const int n0   = blockIdx.x * 64;
  const int wm   = (wave >> 1) * 32;   // wave's M offset in block tile
  const int wn   = (wave & 1) * 32;    // wave's N offset in block tile
  const int row  = lane & 15;
  const int kh16 = (lane >> 4) * 16;   // byte offset for K-half

  const unsigned A_LDS[2] = {0u, 4096u};
  const unsigned B_LDS[2] = {8192u, 12288u};

  // Prologue: DMA the first A/B tiles, publish to all waves.
  if (wave == 0) {
    tdm_load_tile64x32(A_LDS[0], A + (size_t)m0 * K, (unsigned)K);
    tdm_load_tile64x32(B_LDS[0], Bm + (size_t)n0 * K, (unsigned)K);
    __builtin_amdgcn_s_wait_tensorcnt(0);
  }
  __syncthreads();

  v8f acc00 = {}, acc01 = {}, acc10 = {}, acc11 = {};

  for (int k0 = 0; k0 < K; k0 += 32) {
    const int  cur  = (k0 >> 5) & 1;
    const bool more = (k0 + 32) < K;

    // Kick off next tiles' DMA while computing on the current buffer.
    if (wave == 0 && more) {
      tdm_load_tile64x32(A_LDS[cur ^ 1], A + (size_t)m0 * K + (k0 + 32), (unsigned)K);
      tdm_load_tile64x32(B_LDS[cur ^ 1], Bm + (size_t)n0 * K + (k0 + 32), (unsigned)K);
    }

    // Make the TDM's LDS writes visible to the compiler before reading.
    tdm_publish_fence(smem);

    const char* Ab = smem + A_LDS[cur];
    const char* Bb = smem + B_LDS[cur];
    v16bf a0 = lds_frag(Ab + (wm + row) * 64 + kh16);
    v16bf a1 = lds_frag(Ab + (wm + 16 + row) * 64 + kh16);
    v16bf b0 = lds_frag(Bb + (wn + row) * 64 + kh16);
    v16bf b1 = lds_frag(Bb + (wn + 16 + row) * 64 + kh16);

    acc00 = __builtin_amdgcn_wmma_f32_16x16x32_bf16(false, a0, false, b0, (short)0, acc00, false, false);
    acc01 = __builtin_amdgcn_wmma_f32_16x16x32_bf16(false, a0, false, b1, (short)0, acc01, false, false);
    acc10 = __builtin_amdgcn_wmma_f32_16x16x32_bf16(false, a1, false, b0, (short)0, acc10, false, false);
    acc11 = __builtin_amdgcn_wmma_f32_16x16x32_bf16(false, a1, false, b1, (short)0, acc11, false, false);

    if (wave == 0 && more) __builtin_amdgcn_s_wait_tensorcnt(0);
    __syncthreads();
  }

  // C/D layout: lanes 0-15: (M=v, N=lane); lanes 16-31: (M=v+8, N=lane-16)
  const int col  = lane & 15;
  const int mrow = (lane < 16) ? 0 : 8;
#pragma unroll
  for (int v = 0; v < 8; ++v) {
    const size_t r0 = (size_t)(m0 + wm + mrow + v);
    const size_t r1 = (size_t)(m0 + wm + 16 + mrow + v);
    C[r0 * Nc + n0 + wn + col]      = acc00[v];
    C[r0 * Nc + n0 + wn + 16 + col] = acc01[v];
    C[r1 * Nc + n0 + wn + col]      = acc10[v];
    C[r1 * Nc + n0 + wn + 16 + col] = acc11[v];
  }
}

// ---------------------------------------------------------------------------
// Depthwise causal conv (K=4) + bias + SiLU over xs = xr[..., :DI].
// xr is (ROWS, 2*DI). Emits xc (f32) and xc_bf (bf16).
// ---------------------------------------------------------------------------
__global__ void conv_silu_kernel(const float* __restrict__ xr,
                                 const float* __restrict__ conv_w,
                                 const float* __restrict__ conv_b,
                                 float* __restrict__ xc,
                                 __bf16* __restrict__ xc_bf) {
  size_t idx = (size_t)blockIdx.x * blockDim.x + threadIdx.x;
  const size_t total = (size_t)ROWS * DI;
  if (idx >= total) return;
  int d = (int)(idx % DI);
  size_t bl = idx / DI;
  int l = (int)(bl % SEQL);
  int b = (int)(bl / SEQL);

  float acc = conv_b[d];
#pragma unroll
  for (int j = 0; j < KCONV; ++j) {
    int ls = l - (KCONV - 1) + j;
    if (ls >= 0)
      acc += conv_w[d * KCONV + j] * xr[((size_t)b * SEQL + ls) * (2 * DI) + d];
  }
  float s = acc / (1.0f + __expf(-acc));  // silu
  xc[idx]    = s;
  xc_bf[idx] = (__bf16)s;
}

// ---------------------------------------------------------------------------
// B_ssm[row, n] = xc[row, :] . W_xproj[NST + n, :]   (only the B half is used)
// ---------------------------------------------------------------------------
__global__ void xproj_kernel(const float* __restrict__ xc,
                             const float* __restrict__ W_xproj,
                             float* __restrict__ Bssm) {
  int t = blockIdx.x * blockDim.x + threadIdx.x;
  if (t >= ROWS * NST) return;
  int n = t % NST;
  int r = t / NST;
  const float* xrow = xc + (size_t)r * DI;
  const float* wrow = W_xproj + (size_t)(NST + n) * DI;
  float acc = 0.0f;
  for (int d = 0; d < DI; ++d) acc += xrow[d] * wrow[d];
  Bssm[(size_t)r * NST + n] = acc;
}

// ---------------------------------------------------------------------------
// Selective scan. One thread per (b, d) channel; h[16] state in registers.
// delta = softplus(dtraw + b_dt); h = exp(dt*A)*h + (dt*x)*B; y = sum(h).
// Fused epilogue: y = (y + x*D) * silu(res); store bf16 for final GEMM.
// ---------------------------------------------------------------------------
__global__ void scan_kernel(const float* __restrict__ xc,
                            const float* __restrict__ dtraw,
                            const float* __restrict__ Bssm,
                            const float* __restrict__ xr,   // for res half
                            const float* __restrict__ b_dt,
                            const float* __restrict__ A_log,
                            const float* __restrict__ D_param,
                            __bf16* __restrict__ y_bf) {
  int tid = blockIdx.x * blockDim.x + threadIdx.x;
  if (tid >= BSZ * DI) return;
  int b = tid / DI;
  int d = tid % DI;

  float A[NST], h[NST];
#pragma unroll
  for (int n = 0; n < NST; ++n) {
    A[n] = -__expf(A_log[(size_t)d * NST + n]);
    h[n] = 0.0f;
  }
  const float bdt = b_dt[d];
  const float Dp  = D_param[d];

  for (int l = 0; l < SEQL; ++l) {
    size_t base = (size_t)b * SEQL + l;
    float xv = xc[base * DI + d];
    float t  = dtraw[base * DI + d] + bdt;
    float dt = (t > 20.0f) ? t : log1pf(__expf(t));  // softplus
    float dx = dt * xv;
    const float* Bp = Bssm + base * NST;
    float y = 0.0f;
#pragma unroll
    for (int n = 0; n < NST; ++n) {
      float dA = __expf(dt * A[n]);
      h[n] = dA * h[n] + dx * Bp[n];
      y += h[n];
    }
    y += xv * Dp;
    float r    = xr[base * (2 * DI) + DI + d];
    float gate = r / (1.0f + __expf(-r));  // silu(res)
    y_bf[base * DI + d] = (__bf16)(y * gate);
  }
}

// ---------------------------------------------------------------------------
// Launch orchestration
// ---------------------------------------------------------------------------
extern "C" void kernel_launch(void* const* d_in, const int* in_sizes, int n_in,
                              void* d_out, int out_size, void* d_ws, size_t ws_size,
                              hipStream_t stream) {
  const float* x       = (const float*)d_in[0];  // (B, L, Dm)
  const float* W_in    = (const float*)d_in[1];  // (2*Di, Dm)
  const float* conv_w  = (const float*)d_in[2];  // (Di, 1, K)
  const float* conv_b  = (const float*)d_in[3];  // (Di,)
  const float* W_xproj = (const float*)d_in[4];  // (2*N, Di)
  const float* W_dt    = (const float*)d_in[5];  // (Di, Di)
  const float* b_dt    = (const float*)d_in[6];  // (Di,)
  const float* A_log   = (const float*)d_in[7];  // (Di, N)
  const float* D_param = (const float*)d_in[8];  // (Di,)
  const float* W_out   = (const float*)d_in[9];  // (Dm, Di)
  float* out = (float*)d_out;                    // (B, L, Dm)

  // Workspace carve-out (256B aligned slabs).
  char* ws = (char*)d_ws;
  size_t off = 0;
  auto alloc = [&](size_t bytes) -> void* {
    void* p = ws + off;
    off = (off + bytes + 255) & ~(size_t)255;
    return p;
  };
  __bf16* x_bf    = (__bf16*)alloc((size_t)ROWS * DM * 2);        // 8.4 MB
  __bf16* Win_bf  = (__bf16*)alloc((size_t)2 * DI * DM * 2);      // 8.4 MB
  __bf16* Wdt_bf  = (__bf16*)alloc((size_t)DI * DI * 2);          // 8.4 MB
  __bf16* Wout_bf = (__bf16*)alloc((size_t)DM * DI * 2);          // 4.2 MB
  float*  xr      = (float*)alloc((size_t)ROWS * 2 * DI * 4);     // 67 MB
  float*  xc      = (float*)alloc((size_t)ROWS * DI * 4);         // 33.6 MB
  __bf16* xc_bf   = (__bf16*)alloc((size_t)ROWS * DI * 2);        // 16.8 MB
  float*  dtraw   = (float*)alloc((size_t)ROWS * DI * 4);         // 33.6 MB
  float*  Bssm    = (float*)alloc((size_t)ROWS * NST * 4);        // 0.26 MB
  __bf16* y_bf    = (__bf16*)alloc((size_t)ROWS * DI * 2);        // 16.8 MB
  (void)ws_size; (void)in_sizes; (void)n_in; (void)out_size;

  const int CB = 256;
  auto nblk = [](size_t n, int b) { return (unsigned)((n + b - 1) / b); };

  // 1) bf16 copies of GEMM operands.
  cast_f32_bf16<<<nblk((size_t)ROWS * DM, CB), CB, 0, stream>>>(x, x_bf, (size_t)ROWS * DM);
  cast_f32_bf16<<<nblk((size_t)2 * DI * DM, CB), CB, 0, stream>>>(W_in, Win_bf, (size_t)2 * DI * DM);
  cast_f32_bf16<<<nblk((size_t)DI * DI, CB), CB, 0, stream>>>(W_dt, Wdt_bf, (size_t)DI * DI);
  cast_f32_bf16<<<nblk((size_t)DM * DI, CB), CB, 0, stream>>>(W_out, Wout_bf, (size_t)DM * DI);

  // 2) xr = x @ W_in^T : (4096 x 1024) * (4096 x 1024)^T -> (4096 x 4096)
  {
    dim3 grid((2 * DI) / 64, ROWS / 64);
    gemm_bf16_tdm<<<grid, 128, 0, stream>>>(x_bf, Win_bf, xr, ROWS, 2 * DI, DM);
  }

  // 3) depthwise conv + bias + SiLU -> xc (f32 + bf16)
  conv_silu_kernel<<<nblk((size_t)ROWS * DI, CB), CB, 0, stream>>>(xr, conv_w, conv_b, xc, xc_bf);

  // 4) B_ssm projection (tiny)
  xproj_kernel<<<nblk((size_t)ROWS * NST, CB), CB, 0, stream>>>(xc, W_xproj, Bssm);

  // 5) dtraw = xc @ W_dt^T : (4096 x 2048) * (2048 x 2048)^T
  {
    dim3 grid(DI / 64, ROWS / 64);
    gemm_bf16_tdm<<<grid, 128, 0, stream>>>(xc_bf, Wdt_bf, dtraw, ROWS, DI, DI);
  }

  // 6) selective scan + gating epilogue -> y_bf
  scan_kernel<<<nblk((size_t)BSZ * DI, CB), CB, 0, stream>>>(
      xc, dtraw, Bssm, xr, b_dt, A_log, D_param, y_bf);

  // 7) out = y @ W_out^T : (4096 x 2048) * (1024 x 2048)^T -> (4096 x 1024)
  {
    dim3 grid(DM / 64, ROWS / 64);
    gemm_bf16_tdm<<<grid, 128, 0, stream>>>(y_bf, Wout_bf, out, ROWS, DM, DI);
  }
}
